// FasterRCNN_84610855731301
// MI455X (gfx1250) — matrix operand order repaired and checked by reading, not compiled, not run on hardware
//
#include <hip/hip_runtime.h>
#include <hip/hip_bf16.h>

// ---------------- types ----------------
typedef __attribute__((ext_vector_type(16))) __bf16        v16bf;
typedef __attribute__((ext_vector_type(8)))  float         v8f;
typedef __attribute__((ext_vector_type(4))) unsigned int   u32x4;
typedef unsigned short u16;

// ---------------- constants ----------------
#define IMG   800.0f
#define HF    25
#define CF    2048
#define BATCH 2
#define NPOS  625            // 25*25
#define NANC  5625           // 625*9
#define NCOL  1250           // BATCH*NPOS
#define NCOLP 1280           // padded leading dim (multiple of 8 -> 16B aligned rows)
#define KCONV 18432          // 2048*9
#define TOPN  300
#define KFC1  100352         // 2048*49
#define NHID  1024
#define NCLS  81
#define NBOX  320
#define LDOUT 401            // 81+320

#define BM 128
#define BN 128
#define BK 32

__device__ __forceinline__ u16 f2bf(float f) {
    unsigned u = __builtin_bit_cast(unsigned, f);
    unsigned r = u + 0x7FFFu + ((u >> 16) & 1u);   // round to nearest even
    return (u16)(r >> 16);
}
__device__ __forceinline__ float clampf(float v, float lo, float hi) {
    return fminf(fmaxf(v, lo), hi);
}

// ---------------- generic f32 -> bf16 convert ----------------
__global__ void cvt_bf16_kernel(const float* __restrict__ in, u16* __restrict__ out, size_t n) {
    size_t i = (size_t)blockIdx.x * blockDim.x + threadIdx.x;
    if (i < n) out[i] = f2bf(in[i]);
}

// ---------------- im2col for 3x3 SAME conv -> bf16 col (KCONV x NCOL, ld NCOLP) ----------------
__global__ void im2col_kernel(const float* __restrict__ fm, u16* __restrict__ col) {
    size_t idx = (size_t)blockIdx.x * blockDim.x + threadIdx.x;
    if (idx >= (size_t)KCONV * NCOL) return;
    int n    = (int)(idx % NCOL);
    int krow = (int)(idx / NCOL);
    int c = krow / 9, t = krow % 9, dy = t / 3, dx = t % 3;
    int b = n / NPOS, p = n % NPOS, y = p / HF, x = p % HF;
    int yy = y + dy - 1, xx = x + dx - 1;
    float v = 0.0f;
    if (yy >= 0 && yy < HF && xx >= 0 && xx < HF)
        v = fm[((size_t)b * CF + c) * NPOS + yy * HF + xx];
    col[(size_t)krow * NCOLP + n] = f2bf(v);
}

// ---------------- bf16 WMMA GEMM: C(MxN) = A(MxK, ld=lda) * B(KxN, ld=ldb) ----------------
// Fragment layouts follow CDNA5 ISA 7.12.2 (16-bit A 16x32, B 32x16, f32 C 16x16).
// Split-K: blockIdx.z selects K-slice [z*Kc, z*Kc+Kc); partials -> Cpart[z][M][ldc].
struct P16 { union { struct { u32x4 q0, q1; } v; u16 e[16]; } u; };

__device__ __forceinline__ v16bf load_frag_2x16B(const u16* p0, const u16* p1) {
    union { u32x4 q[2]; v16bf v; } f;
    f.q[0] = *(const u32x4*)p0;
    f.q[1] = *(const u32x4*)p1;
    return f.v;
}

__global__ __launch_bounds__(256) void gemm_bf16_kernel(
    const u16* __restrict__ A, int lda, const u16* __restrict__ B, int ldb,
    float* __restrict__ Cf, u16* __restrict__ Cbf,
    const float* __restrict__ bias, int biasPerRow, int relu,
    int M, int N, int K, int ldc, int Kc, float* __restrict__ Cpart)
{
    __shared__ __attribute__((aligned(16))) u16 aS[BM * BK];  // [m][k]
    __shared__ __attribute__((aligned(16))) u16 bS[BN * BK];  // transposed: [n][k]

    const int tid  = threadIdx.x;
    const int lane = tid & 31;
    const int wave = tid >> 5;          // 0..7
    const int wm   = wave >> 2;         // 0..1  -> 64 rows
    const int wn   = wave & 3;          // 0..3  -> 32 cols
    const int m0   = blockIdx.y * BM;
    const int n0   = blockIdx.x * BN;
    const int l16  = lane & 15;
    const int half = lane >> 4;         // 0/1
    const int kbA  = half * 8;          // A frag: k elems {kbA..kbA+7, kbA+16..kbA+23}
    const int kbB  = half * 8;          // B frag: k elems {kbB..kbB+7, kbB+16..kbB+23}

    const int kB = blockIdx.z * Kc;
    const int kE = (kB + Kc < K) ? (kB + Kc) : K;

    const bool laOK = ((lda & 7) == 0);
    const bool lbOK = ((ldb & 7) == 0);

    // staging coordinates
    const int amS = tid >> 1;            // A: row in tile
    const int akS = (tid & 1) * 16;      // A: k offset
    const int bkS = tid & 31;            // B: k in tile (one k per lane -> stride-1 LDS banking)
    const int bnS = (tid >> 5) * 16;     // B: n offset (16 cols per thread)

    auto loadA = [&](int k0) -> P16 {
        P16 r;
        int gm = m0 + amS, gk = k0 + akS;
        if (laOK && gm < M && gk + 16 <= K) {
            const u16* src = &A[(size_t)gm * lda + gk];
            r.u.v.q0 = ((const u32x4*)src)[0];
            r.u.v.q1 = ((const u32x4*)src)[1];
            if (k0 + BK < kE) __builtin_prefetch(&A[(size_t)gm * lda + gk + BK], 0, 0);
        } else {
            #pragma unroll
            for (int i = 0; i < 16; ++i) {
                int g = gk + i;
                r.u.e[i] = (gm < M && g < K) ? A[(size_t)gm * lda + g] : (u16)0;
            }
        }
        return r;
    };
    auto loadB = [&](int k0) -> P16 {
        P16 r;
        int gk = k0 + bkS, gn = n0 + bnS;
        if (lbOK && gk < K && gn + 16 <= N) {
            const u16* src = &B[(size_t)gk * ldb + gn];
            r.u.v.q0 = ((const u32x4*)src)[0];
            r.u.v.q1 = ((const u32x4*)src)[1];
        } else {
            #pragma unroll
            for (int i = 0; i < 16; ++i) {
                int g = gn + i;
                r.u.e[i] = (gk < K && g < N) ? B[(size_t)gk * ldb + g] : (u16)0;
            }
        }
        return r;
    };

    // fragment LDS base pointers (k0-invariant)
    const u16* aP[4];
    #pragma unroll
    for (int i = 0; i < 4; ++i) aP[i] = &aS[(wm * 64 + i * 16 + l16) * BK + kbA];
    const u16* bP[2];
    #pragma unroll
    for (int j = 0; j < 2; ++j) bP[j] = &bS[(wn * 32 + j * 16 + l16) * BK + kbB];

    v8f acc[4][2];
    #pragma unroll
    for (int i = 0; i < 4; ++i)
        #pragma unroll
        for (int j = 0; j < 2; ++j)
            acc[i][j] = (v8f){0.f,0.f,0.f,0.f,0.f,0.f,0.f,0.f};

    P16 pa = loadA(kB), pb = loadB(kB);

    for (int k0 = kB; k0 < kE; k0 += BK) {
        // ---- commit staged registers to LDS ----
        {
            u32x4* d = (u32x4*)&aS[amS * BK + akS];
            d[0] = pa.u.v.q0; d[1] = pa.u.v.q1;
        }
        {
            #pragma unroll
            for (int i = 0; i < 16; ++i) bS[(bnS + i) * BK + bkS] = pb.u.e[i];
        }
        __syncthreads();

        // ---- software pipeline: issue next-tile global loads before compute ----
        const bool more = (k0 + BK) < kE;
        P16 na = pa, nb = pb;
        if (more) { na = loadA(k0 + BK); nb = loadB(k0 + BK); }

        // ---- B fragments (2 per wave) ----
        v16bf bfrag[2];
        #pragma unroll
        for (int j = 0; j < 2; ++j) bfrag[j] = load_frag_2x16B(bP[j], bP[j] + 16);

        // ---- 4 A fragments x 2 B fragments -> 8 WMMAs ----
        #pragma unroll
        for (int i = 0; i < 4; ++i) {
            v16bf afrag = load_frag_2x16B(aP[i], aP[i] + 16);
            #pragma unroll
            for (int j = 0; j < 2; ++j) {
                acc[i][j] = __builtin_amdgcn_wmma_f32_16x16x32_bf16(
                    false, afrag, false, bfrag[j], (short)0, acc[i][j], false, false);
            }
        }
        __syncthreads();
        pa = na; pb = nb;
    }

    // ---- epilogue ----
    #pragma unroll
    for (int i = 0; i < 4; ++i) {
        #pragma unroll
        for (int j = 0; j < 2; ++j) {
            int nIdx  = n0 + wn * 32 + j * 16 + l16;
            int mBase = m0 + wm * 64 + i * 16 + half * 8;
            if (nIdx >= N) continue;
            #pragma unroll
            for (int r = 0; r < 8; ++r) {
                int mIdx = mBase + r;
                if (mIdx >= M) continue;
                float v = acc[i][j][r];
                if (Cpart) {   // split-K: raw partial sums, fixed slice slot
                    Cpart[((size_t)blockIdx.z * M + mIdx) * ldc + nIdx] = v;
                } else {
                    if (bias) v += biasPerRow ? bias[mIdx] : bias[nIdx];
                    if (relu) v = fmaxf(v, 0.0f);
                    if (Cf)  Cf [(size_t)mIdx * ldc + nIdx] = v;
                    if (Cbf) Cbf[(size_t)mIdx * ldc + nIdx] = f2bf(v);
                }
            }
        }
    }
}

// ---------------- deterministic split-K reduction + bias/relu ----------------
__global__ void reduce_split_kernel(const float* __restrict__ part, int S, int M, int N, int ldp,
                                    const float* __restrict__ bias, int biasPerRow, int relu,
                                    float* __restrict__ Cf, u16* __restrict__ Cbf, int ldc)
{
    size_t idx = (size_t)blockIdx.x * blockDim.x + threadIdx.x;
    if (idx >= (size_t)M * N) return;
    int m = (int)(idx / N), n = (int)(idx % N);
    float v = 0.0f;
    for (int s = 0; s < S; ++s) v += part[((size_t)s * M + m) * ldp + n];   // fixed order
    if (bias) v += biasPerRow ? bias[m] : bias[n];
    if (relu) v = fmaxf(v, 0.0f);
    if (Cf)  Cf [(size_t)m * ldc + n] = v;
    if (Cbf) Cbf[(size_t)m * ldc + n] = f2bf(v);
}

// ---------------- proposals: sigmoid score + anchor decode + clip ----------------
__global__ void proposal_kernel(const float* __restrict__ cls, const float* __restrict__ bbox,
                                float* __restrict__ props, float* __restrict__ scores)
{
    int idx = blockIdx.x * blockDim.x + threadIdx.x;
    if (idx >= BATCH * NANC) return;
    int b = idx / NANC, i = idx % NANC;
    int p = i / 9, a = i % 9;
    int col = b * NPOS + p;

    float l0 = cls[(size_t)(2 * a    ) * NCOLP + col];
    float l1 = cls[(size_t)(2 * a + 1) * NCOLP + col];
    float sc = 1.0f / (1.0f + expf(l0 - l1));   // softmax[...,1]

    float d0 = bbox[(size_t)(4 * a + 0) * NCOLP + col];
    float d1 = bbox[(size_t)(4 * a + 1) * NCOLP + col];
    float d2 = bbox[(size_t)(4 * a + 2) * NCOLP + col];
    float d3 = bbox[(size_t)(4 * a + 3) * NCOLP + col];

    int y = p / HF, x = p % HF;
    const float SC[3] = {64.0f, 128.0f, 256.0f};
    const float SQ[3] = {0.70710678118654752f, 1.0f, 1.41421356237309505f};
    int si = a / 3, ri = a % 3;
    float aw = SC[si] * SQ[ri];
    float ah = SC[si] / SQ[ri];
    float cx = (x + 0.5f) * 32.0f, cy = (y + 0.5f) * 32.0f;
    float ax1 = clampf(cx - 0.5f * aw, 0.f, IMG);
    float ay1 = clampf(cy - 0.5f * ah, 0.f, IMG);
    float ax2 = clampf(cx + 0.5f * aw, 0.f, IMG);
    float ay2 = clampf(cy + 0.5f * ah, 0.f, IMG);

    float w = ax2 - ax1, h = ay2 - ay1;
    float ccx = ax1 + 0.5f * w, ccy = ay1 + 0.5f * h;
    float pcx = ccx + d0 * w,   pcy = ccy + d1 * h;
    float pw  = w * expf(d2),   ph  = h * expf(d3);

    float* o = &props[(size_t)idx * 4];
    o[0] = clampf(pcx - 0.5f * pw, 0.f, IMG);
    o[1] = clampf(pcy - 0.5f * ph, 0.f, IMG);
    o[2] = clampf(pcx + 0.5f * pw, 0.f, IMG);
    o[3] = clampf(pcy + 0.5f * ph, 0.f, IMG);
    scores[idx] = sc;
}

// ---------------- greedy NMS, fused suppress+argmax scan, wave32 shuffle reduce ----------------
__global__ __launch_bounds__(256) void nms_kernel(const float* __restrict__ props,
                                                  const float* __restrict__ scores,
                                                  float* __restrict__ rois)
{
    int b = blockIdx.x;
    const float* P = props  + (size_t)b * NANC * 4;
    const float* S = scores + (size_t)b * NANC;
    __shared__ float sc[NANC];
    __shared__ float wv[8];
    __shared__ int   wi[8];
    __shared__ int   s_best, s_first;
    __shared__ float s_val;
    __shared__ float s_box[4];
    int tid = threadIdx.x, lane = tid & 31, wid = tid >> 5;

    for (int i = tid; i < NANC; i += 256) sc[i] = S[i];   // all > 0 (sigmoid)
    __syncthreads();

    // ---- initial argmax (smallest-index tie break == stable descending argsort) ----
    float bv = -1.0f; int bi = NANC;
    for (int i = tid; i < NANC; i += 256) { float v = sc[i]; if (v > bv) { bv = v; bi = i; } }
    #pragma unroll
    for (int d = 16; d > 0; d >>= 1) {
        float ov = __shfl_xor(bv, d, 32);
        int   oi = __shfl_xor(bi, d, 32);
        if (ov > bv || (ov == bv && oi < bi)) { bv = ov; bi = oi; }
    }
    if (lane == 0) { wv[wid] = bv; wi[wid] = bi; }
    __syncthreads();
    if (tid == 0) {
        float fv = wv[0]; int fi = wi[0];
        for (int w = 1; w < 8; ++w)
            if (wv[w] > fv || (wv[w] == fv && wi[w] < fi)) { fv = wv[w]; fi = wi[w]; }
        s_first = fi; s_best = fi; s_val = fv;
        s_box[0] = P[fi*4+0]; s_box[1] = P[fi*4+1]; s_box[2] = P[fi*4+2]; s_box[3] = P[fi*4+3];
    }
    __syncthreads();

    for (int t = 0; t < TOPN; ++t) {
        const int   best = s_best;
        const float bval = s_val;
        const float x1 = s_box[0], y1 = s_box[1], x2 = s_box[2], y2 = s_box[3];
        if (tid == 0) {
            float* r = &rois[((size_t)b * TOPN + t) * 4];
            r[0] = x1; r[1] = y1; r[2] = x2; r[3] = y2;
        }
        // fused: suppress vs current pick while scanning for the next max
        const float area1 = (x2 - x1) * (y2 - y1);
        const bool  live  = (bval >= 0.0f);
        bv = -1.0f; bi = NANC;
        for (int i = tid; i < NANC; i += 256) {
            float v = sc[i];
            if (v < 0.0f) continue;
            if (live) {
                if (i == best) { sc[i] = -2.0f; continue; }
                float bx1 = P[i*4+0], by1 = P[i*4+1], bx2 = P[i*4+2], by2 = P[i*4+3];
                float ix1 = fmaxf(x1, bx1), iy1 = fmaxf(y1, by1);
                float ix2 = fminf(x2, bx2), iy2 = fminf(y2, by2);
                float iw = ix2 - ix1, ih = iy2 - iy1;
                float inter = (iw > 0.f && ih > 0.f) ? iw * ih : 0.f;
                float a2 = (bx2 - bx1) * (by2 - by1);
                float iou = inter / (area1 + a2 - inter);
                if (iou > 0.7f) { sc[i] = -2.0f; continue; }
            }
            if (v > bv) { bv = v; bi = i; }
        }
        #pragma unroll
        for (int d = 16; d > 0; d >>= 1) {
            float ov = __shfl_xor(bv, d, 32);
            int   oi = __shfl_xor(bi, d, 32);
            if (ov > bv || (ov == bv && oi < bi)) { bv = ov; bi = oi; }
        }
        if (lane == 0) { wv[wid] = bv; wi[wid] = bi; }
        __syncthreads();
        if (tid == 0) {
            float fv = wv[0]; int fi = wi[0];
            for (int w = 1; w < 8; ++w)
                if (wv[w] > fv || (wv[w] == fv && wi[w] < fi)) { fv = wv[w]; fi = wi[w]; }
            if (fv < 0.0f) fi = s_first;    // exhausted -> pad with order[0]
            s_best = fi; s_val = fv;
            s_box[0] = P[fi*4+0]; s_box[1] = P[fi*4+1]; s_box[2] = P[fi*4+2]; s_box[3] = P[fi*4+3];
        }
        __syncthreads();
    }
}

// ---------------- ROI align -> flattened bf16 rows (600 x 100352) ----------------
__global__ void roi_align_kernel(const float* __restrict__ fm, const float* __restrict__ rois,
                                 u16* __restrict__ outBf)
{
    size_t idx = (size_t)blockIdx.x * blockDim.x + threadIdx.x;
    const size_t total = (size_t)BATCH * TOPN * CF * 49;
    if (idx >= total) return;
    int s   = (int)(idx % 49);
    int c   = (int)((idx / 49) % CF);
    int row = (int)(idx / ((size_t)49 * CF));
    int b   = row / TOPN;
    int oy = s / 7, ox = s % 7;

    const float* R = &rois[(size_t)row * 4];
    const float inv = 1.0f / 32.0f;
    float x1 = R[0] * inv, y1 = R[1] * inv, x2 = R[2] * inv, y2 = R[3] * inv;
    float gx = x1 + (x2 - x1) * (ox * (1.0f / 6.0f));
    float gy = y1 + (y2 - y1) * (oy * (1.0f / 6.0f));
    float fx0 = floorf(gx), fy0 = floorf(gy);
    float wx = gx - fx0, wy = gy - fy0;
    int x0 = (int)fx0, y0 = (int)fy0;

    const float* F = fm + ((size_t)b * CF + c) * NPOS;
    auto samp = [&](int yi, int xi) -> float {
        bool valid = (yi >= 0) && (yi < HF) && (xi >= 0) && (xi < HF);
        int yc = yi < 0 ? 0 : (yi > HF - 1 ? HF - 1 : yi);
        int xc = xi < 0 ? 0 : (xi > HF - 1 ? HF - 1 : xi);
        float v = F[yc * HF + xc];
        return valid ? v : 0.0f;
    };
    float v = samp(y0,     x0    ) * (1.f - wy) * (1.f - wx)
            + samp(y0,     x0 + 1) * (1.f - wy) * wx
            + samp(y0 + 1, x0    ) * wy * (1.f - wx)
            + samp(y0 + 1, x0 + 1) * wy * wx;
    outBf[idx] = f2bf(v);
}

// ---------------- host launch ----------------
extern "C" void kernel_launch(void* const* d_in, const int* in_sizes, int n_in,
                              void* d_out, int out_size, void* d_ws, size_t ws_size,
                              hipStream_t stream) {
    (void)in_sizes; (void)n_in; (void)out_size; (void)ws_size;
    const float* fm        = (const float*)d_in[0];
    const float* rpn_w     = (const float*)d_in[1];
    const float* rpn_b     = (const float*)d_in[2];
    const float* rpn_cls_w = (const float*)d_in[3];
    const float* rpn_cls_b = (const float*)d_in[4];
    const float* rpn_bbx_w = (const float*)d_in[5];
    const float* rpn_bbx_b = (const float*)d_in[6];
    const float* fc1_w     = (const float*)d_in[7];
    const float* fc1_b     = (const float*)d_in[8];
    const float* fc2_w     = (const float*)d_in[9];
    const float* fc2_b     = (const float*)d_in[10];
    const float* cls_w     = (const float*)d_in[11];
    const float* cls_b     = (const float*)d_in[12];
    const float* bbox_w    = (const float*)d_in[13];
    const float* bbox_b    = (const float*)d_in[14];
    float* out = (float*)d_out;

    char* ws = (char*)d_ws;
    size_t off = 0;
    auto alloc = [&](size_t bytes) -> char* {
        char* p = ws + off;
        off = (off + bytes + 255) & ~(size_t)255;
        return p;
    };
    u16*   rpnW_bf  = (u16*)alloc((size_t)512 * KCONV * 2);
    u16*   clsW_bf  = (u16*)alloc((size_t)18 * 512 * 2);
    u16*   bbxW_bf  = (u16*)alloc((size_t)36 * 512 * 2);
    u16*   fc1W_bf  = (u16*)alloc((size_t)KFC1 * NHID * 2);
    u16*   fc2W_bf  = (u16*)alloc((size_t)NHID * NHID * 2);
    u16*   clW_bf   = (u16*)alloc((size_t)NHID * NCLS * 2);
    u16*   bbW_bf   = (u16*)alloc((size_t)NHID * NBOX * 2);
    u16*   col_bf   = (u16*)alloc((size_t)KCONV * NCOLP * 2);
    u16*   x_bf     = (u16*)alloc((size_t)512 * NCOLP * 2);
    float* clsF     = (float*)alloc((size_t)18 * NCOLP * 4);
    float* bbxF     = (float*)alloc((size_t)36 * NCOLP * 4);
    float* props    = (float*)alloc((size_t)BATCH * NANC * 4 * 4);
    float* scores   = (float*)alloc((size_t)BATCH * NANC * 4);
    float* rois     = (float*)alloc((size_t)BATCH * TOPN * 4 * 4);
    u16*   roi_bf   = (u16*)alloc((size_t)BATCH * TOPN * KFC1 * 2);
    u16*   h1_bf    = (u16*)alloc((size_t)BATCH * TOPN * NHID * 2);
    u16*   h2_bf    = (u16*)alloc((size_t)BATCH * TOPN * NHID * 2);
    float* partials = (float*)alloc((size_t)8 * 512 * NCOLP * 4);   // >= all split-K uses

    auto cvt = [&](const float* src, u16* dst, size_t n) {
        cvt_bf16_kernel<<<(unsigned)((n + 255) / 256), 256, 0, stream>>>(src, dst, n);
    };
    cvt(rpn_w,     rpnW_bf, (size_t)512 * KCONV);
    cvt(rpn_cls_w, clsW_bf, (size_t)18 * 512);
    cvt(rpn_bbx_w, bbxW_bf, (size_t)36 * 512);
    cvt(fc1_w,     fc1W_bf, (size_t)KFC1 * NHID);
    cvt(fc2_w,     fc2W_bf, (size_t)NHID * NHID);
    cvt(cls_w,     clW_bf,  (size_t)NHID * NCLS);
    cvt(bbox_w,    bbW_bf,  (size_t)NHID * NBOX);

    {   // im2col
        size_t n = (size_t)KCONV * NCOL;
        im2col_kernel<<<(unsigned)((n + 255) / 256), 256, 0, stream>>>(fm, col_bf);
    }
    // S must divide K into BK-multiples (all call sites verified: K/S % 32 == 0)
    auto gemm = [&](const u16* A, int lda, const u16* B, int ldb,
                    float* Cf, u16* Cbf, const float* bias,
                    int biasPerRow, int relu, int M, int N, int K, int ldc, int S) {
        dim3 grid((N + BN - 1) / BN, (M + BM - 1) / BM, S);
        int Kc = K / S;
        if (S > 1) {
            gemm_bf16_kernel<<<grid, 256, 0, stream>>>(A, lda, B, ldb, nullptr, nullptr,
                                                       nullptr, 0, 0, M, N, K, ldc, Kc, partials);
            size_t tot = (size_t)M * N;
            reduce_split_kernel<<<(unsigned)((tot + 255) / 256), 256, 0, stream>>>(
                partials, S, M, N, ldc, bias, biasPerRow, relu, Cf, Cbf, ldc);
        } else {
            gemm_bf16_kernel<<<grid, 256, 0, stream>>>(A, lda, B, ldb, Cf, Cbf, bias,
                                                       biasPerRow, relu, M, N, K, ldc, K, nullptr);
        }
    };
    // RPN 3x3 conv + ReLU:  x(512 x 1250, ld 1280) bf16   (split-K x8 -> 320 WGs)
    gemm(rpnW_bf, KCONV, col_bf, NCOLP, nullptr, x_bf, rpn_b,     1, 1, 512, NCOL, KCONV, NCOLP, 8);
    // RPN heads (1x1): logits f32 (ld 1280)                (split-K x4)
    gemm(clsW_bf, 512,   x_bf,   NCOLP, clsF, nullptr, rpn_cls_b, 1, 0, 18,  NCOL, 512, NCOLP, 4);
    gemm(bbxW_bf, 512,   x_bf,   NCOLP, bbxF, nullptr, rpn_bbx_b, 1, 0, 36,  NCOL, 512, NCOLP, 4);

    proposal_kernel<<<(BATCH * NANC + 255) / 256, 256, 0, stream>>>(clsF, bbxF, props, scores);
    nms_kernel<<<BATCH, 256, 0, stream>>>(props, scores, rois);

    {   // ROI align -> bf16 flattened rows
        size_t n = (size_t)BATCH * TOPN * CF * 49;
        roi_align_kernel<<<(unsigned)((n + 255) / 256), 256, 0, stream>>>(fm, rois, roi_bf);
    }
    // FC head: fc1 split-K x8 -> 320 WGs, fc2 split-K x4 -> 160 WGs
    gemm(roi_bf, KFC1, fc1W_bf, NHID, nullptr, h1_bf, fc1_b,  0, 1, BATCH * TOPN, NHID, KFC1, NHID, 8);
    gemm(h1_bf,  NHID, fc2W_bf, NHID, nullptr, h2_bf, fc2_b,  0, 1, BATCH * TOPN, NHID, NHID, NHID, 4);
    // Final outputs straight into d_out (row stride 401)
    gemm(h2_bf,  NHID, clW_bf,  NCLS, out,      nullptr, cls_b,  0, 0, BATCH * TOPN, NCLS, NHID, LDOUT, 1);
    gemm(h2_bf,  NHID, bbW_bf,  NBOX, out + 81, nullptr, bbox_b, 0, 0, BATCH * TOPN, NBOX, NHID, LDOUT, 1);
}
